// QNetworkSwapGNN_4604204941422
// MI455X (gfx1250) — compile-verified, matrix operand order checked by reading, not run on hardware
//
#include <hip/hip_runtime.h>
#include <hip/hip_bf16.h>

// ---------------------------------------------------------------------------
// QNetworkSwapGNN for MI455X (gfx1250): WMMA bf16 GEMMs + atomic edge scatter
// ---------------------------------------------------------------------------

#define N_NODES 50000
#define N_EDGES 800000

typedef __bf16 bf16;
typedef __attribute__((ext_vector_type(8)))  __bf16 v8bf;
typedef __attribute__((ext_vector_type(16))) __bf16 v16bf;
typedef __attribute__((ext_vector_type(8)))  float  v8f;

union V16U { v16bf v; v8bf h[2]; };

__device__ __forceinline__ bf16 f2bf(float f) {
    union { float f; unsigned u; } c; c.f = f;
    unsigned r = (c.u + 0x7FFFu + ((c.u >> 16) & 1u)) >> 16;
    union { unsigned short s; bf16 b; } o; o.s = (unsigned short)r;
    return o.b;
}

// monotone float<->uint mapping so unsigned atomicMax == float max
__device__ __forceinline__ unsigned fmap(float f) {
    unsigned b = __float_as_uint(f);
    return (b & 0x80000000u) ? ~b : (b | 0x80000000u);
}
__device__ __forceinline__ float funmap(unsigned u) {
    return __uint_as_float((u & 0x80000000u) ? (u & 0x7FFFFFFFu) : ~u);
}

__device__ __forceinline__ float rm_of(float m, int i) {
    bool loc = i < 15;
    if (loc && m == 0.0f)                return -__builtin_inff();
    if (loc && m == -__builtin_inff())   return 0.0f;
    return m;
}

// ---------------------------------------------------------------------------
// Weight packing into WMMA B-fragment layout (ISA 7.12.2, 16-bit operands):
// lane L holds col n = L%16;   element j -> k = kt*32 + (j&7) + 8*(L>>4) + 16*(j>>3)
// Fragment f = kt*ntStride + nt, stored as Bp[f*512 + lane*16 + j].
// ---------------------------------------------------------------------------
__global__ __launch_bounds__(256)
void pack_b(const float* __restrict__ W, bf16* __restrict__ Bp,
            int ktCount, int ntBase, int ntCount, int ntStride,
            int actualK, int rowlen) {
    int tid = blockIdx.x * blockDim.x + threadIdx.x;
    int total = ktCount * ntCount * 512;
    if (tid >= total) return;
    int j    = tid & 15;
    int lane = (tid >> 4) & 31;
    int rest = tid >> 9;
    int ntL  = rest % ntCount;
    int kt   = rest / ntCount;
    int nLoc = ntL * 16 + (lane & 15);
    int k    = kt * 32 + (j & 7) + 8 * (lane >> 4) + 16 * (j >> 3);
    float v  = (k < actualK && nLoc < rowlen) ? W[(size_t)k * rowlen + nLoc] : 0.0f;
    Bp[((size_t)(kt * ntStride + ntBase + ntL) * 32 + lane) * 16 + j] = f2bf(v);
}

__global__ void concat_bias(const float* __restrict__ a, const float* __restrict__ b,
                            const float* __restrict__ c, const float* __restrict__ d,
                            float* __restrict__ out) {
    int i = blockIdx.x * blockDim.x + threadIdx.x;
    if (i >= 512) return;
    const float* p[4] = {a, b, c, d};
    out[i] = p[i >> 7][i & 127];
}

// ---------------------------------------------------------------------------
// Generic WMMA GEMM: C[N x storeCols] = act(A[N x K] @ B + bias)
// One wave per 16x16 output tile; K multiple of 32; N multiple of 16 (50000 ok)
// A fragment per lane: k-halves at A[(row0+L%16)*K + kt*32 + 8*(L>>4) + {0,16}]
// ---------------------------------------------------------------------------
__global__ __launch_bounds__(256)
void gemm_wmma(const bf16* __restrict__ A, int K,
               const bf16* __restrict__ Bp,
               const float* __restrict__ bias,
               float* __restrict__ Cf, bf16* __restrict__ Cb,
               int Nrows, int NC, int storeCols, int relu) {
    int wave = (int)((blockIdx.x * blockDim.x + threadIdx.x) >> 5);
    int lane = threadIdx.x & 31;
    int ntiles = NC >> 4;
    int totalTiles = (Nrows >> 4) * ntiles;
    if (wave >= totalTiles) return;
    int rt = wave / ntiles, nt = wave % ntiles;
    int row0  = rt << 4;
    int m     = lane & 15;
    int khalf = lane >> 4;

    v8f acc = {};
    int ktCount = K >> 5;
    for (int kt = 0; kt < ktCount; ++kt) {
        const bf16* arow = A + (size_t)(row0 + m) * K + kt * 32 + khalf * 8;
        V16U a, b;
        a.h[0] = *(const v8bf*)(arow);
        a.h[1] = *(const v8bf*)(arow + 16);
        const bf16* bp = Bp + ((size_t)(kt * ntiles + nt) * 32 + lane) * 16;
        b.h[0] = *(const v8bf*)(bp);
        b.h[1] = *(const v8bf*)(bp + 8);
        acc = __builtin_amdgcn_wmma_f32_16x16x32_bf16(false, a.v, false, b.v,
                                                      (short)0, acc, false, false);
    }

    int n = nt * 16 + (lane & 15);
    if (n < storeCols) {
        float bv = bias ? bias[n] : 0.0f;
#pragma unroll
        for (int r = 0; r < 8; ++r) {
            int mm = khalf * 8 + r;
            float val = acc[r] + bv;
            if (relu) val = fmaxf(val, 0.0f);
            size_t idx = (size_t)(row0 + mm) * storeCols + n;
            if (Cf) Cf[idx] = val;
            if (Cb) Cb[idx] = f2bf(val);
        }
    }
}

// ---------------------------------------------------------------------------
// Node feature assembly: x0 = [node_type, requests, update_step, 0...0] (N x 32)
// ---------------------------------------------------------------------------
__global__ __launch_bounds__(256)
void build_x0(const float* __restrict__ nt, const float* __restrict__ rq,
              const float* __restrict__ us, bf16* __restrict__ XBF) {
    int i = blockIdx.x * blockDim.x + threadIdx.x;
    if (i >= N_NODES * 32) return;
    int n = i >> 5, c = i & 31;
    float v = (c == 0) ? nt[n] : (c == 1) ? rq[n] : (c == 2) ? us[n] : 0.0f;
    XBF[i] = f2bf(v);
}

// ---------------------------------------------------------------------------
// Edge phase (thread per (edge, head)); qkvs row stride 512: q|k|v|s blocks
// ---------------------------------------------------------------------------
__global__ __launch_bounds__(256)
void edge_alpha(const int* __restrict__ ei, const float* __restrict__ lat,
                const float* __restrict__ qkvs,
                const float* __restrict__ We, const float* __restrict__ be,
                float* __restrict__ alpha, unsigned* __restrict__ amax) {
    int t = blockIdx.x * blockDim.x + threadIdx.x;
    if (t >= N_EDGES * 4) return;
    int e = t >> 2, h = t & 3;
    int src = ei[e], dst = ei[N_EDGES + e];
    const float* q = qkvs + (size_t)dst * 512 + h * 32;
    const float* k = qkvs + (size_t)src * 512 + 128 + h * 32;
    __builtin_prefetch(k, 0, 1);
    float l = lat[e];
    float s = 0.0f;
#pragma unroll
    for (int c = 0; c < 32; ++c)
        s += q[c] * (k[c] + l * We[h * 32 + c] + be[h * 32 + c]);
    s *= 0.17677669529663687f;   // 1/sqrt(EMB)
    alpha[t] = s;
    atomicMax(amax + (size_t)dst * 4 + h, fmap(s));
}

__global__ __launch_bounds__(256)
void edge_exp(const int* __restrict__ ei, const unsigned* __restrict__ amax,
              float* __restrict__ alpha, float* __restrict__ den) {
    int t = blockIdx.x * blockDim.x + threadIdx.x;
    if (t >= N_EDGES * 4) return;
    int e = t >> 2, h = t & 3;
    int dst = ei[N_EDGES + e];
    float mx = funmap(amax[(size_t)dst * 4 + h]);
    float ex = __expf(alpha[t] - mx);
    alpha[t] = ex;
    atomicAdd(den + (size_t)dst * 4 + h, ex);
}

__global__ __launch_bounds__(256)
void edge_msg(const int* __restrict__ ei, const float* __restrict__ lat,
              const float* __restrict__ qkvs,
              const float* __restrict__ We, const float* __restrict__ be,
              const float* __restrict__ alpha, const float* __restrict__ den,
              float* __restrict__ outagg) {
    int t = blockIdx.x * blockDim.x + threadIdx.x;
    if (t >= N_EDGES * 4) return;
    int e = t >> 2, h = t & 3;
    int src = ei[e], dst = ei[N_EDGES + e];
    float a = alpha[t] / (den[(size_t)dst * 4 + h] + 1e-16f);
    float l = lat[e];
    const float* v = qkvs + (size_t)src * 512 + 256 + h * 32;
    __builtin_prefetch(v, 0, 1);
    float* o = outagg + (size_t)dst * 128 + h * 32;
#pragma unroll
    for (int c = 0; c < 32; ++c)
        atomicAdd(o + c, (v[c] + l * We[h * 32 + c] + be[h * 32 + c]) * a);
}

// beta-gated combine: comb = beta*xr + (1-beta)*out,  beta = sigmoid([out|xr|out-xr] @ Wb)
__global__ __launch_bounds__(256)
void gate(const float* __restrict__ qkvs, const float* __restrict__ outagg,
          const float* __restrict__ Wb, bf16* __restrict__ comb) {
    int n = blockIdx.x * blockDim.x + threadIdx.x;
    if (n >= N_NODES) return;
    const float* o  = outagg + (size_t)n * 128;
    const float* xr = qkvs + (size_t)n * 512 + 384;
    float z = 0.0f;
    for (int c = 0; c < 128; ++c) {
        float oc = o[c], xc = xr[c];
        z += oc * Wb[c] + xc * Wb[128 + c] + (oc - xc) * Wb[256 + c];
    }
    float beta = 1.0f / (1.0f + __expf(-z));
    for (int c = 0; c < 128; ++c)
        comb[(size_t)n * 128 + c] = f2bf(beta * xr[c] + (1.0f - beta) * o[c]);
}

// ---------------------------------------------------------------------------
// BatchNorm (population stats over N)
// ---------------------------------------------------------------------------
__global__ __launch_bounds__(256)
void bn_stats(const float* __restrict__ X, float* __restrict__ stat) {
    __shared__ float ssum[256], ssq[256];
    int c = blockIdx.x;
    float s = 0.0f, q = 0.0f;
    for (int i = threadIdx.x; i < N_NODES; i += 256) {
        float v = X[(size_t)i * 32 + c];
        s += v; q += v * v;
    }
    ssum[threadIdx.x] = s; ssq[threadIdx.x] = q;
    __syncthreads();
    for (int st = 128; st > 0; st >>= 1) {
        if (threadIdx.x < st) {
            ssum[threadIdx.x] += ssum[threadIdx.x + st];
            ssq[threadIdx.x]  += ssq[threadIdx.x + st];
        }
        __syncthreads();
    }
    if (threadIdx.x == 0) {
        float m = ssum[0] / (float)N_NODES;
        stat[c]      = m;
        stat[32 + c] = ssq[0] / (float)N_NODES - m * m;
    }
}

__global__ __launch_bounds__(256)
void bn_apply(const float* __restrict__ X, const float* __restrict__ stat,
              const float* __restrict__ g, const float* __restrict__ b,
              bf16* __restrict__ XBF) {
    int i = blockIdx.x * blockDim.x + threadIdx.x;
    if (i >= N_NODES * 32) return;
    int c = i & 31;
    float m = stat[c], v = stat[32 + c];
    XBF[i] = f2bf((X[i] - m) * rsqrtf(v + 1e-5f) * g[c] + b[c]);
}

// ---------------------------------------------------------------------------
// argmax(q[:,0] + rm) with first-index tie-breaking, then final output
// ---------------------------------------------------------------------------
__global__ __launch_bounds__(256)
void argmax_part(const float* __restrict__ QOUT, const float* __restrict__ mask,
                 float* __restrict__ pScore, int* __restrict__ pIdx) {
    __shared__ float sb[256]; __shared__ int si[256];
    float best = -__builtin_inff(); int bi = 0x7FFFFFFF;
    for (int i = blockIdx.x * blockDim.x + threadIdx.x; i < N_NODES;
         i += gridDim.x * blockDim.x) {
        float s = QOUT[(size_t)i * 2] + rm_of(mask[i], i);
        if (s > best || (s == best && i < bi)) { best = s; bi = i; }
    }
    int t = threadIdx.x;
    sb[t] = best; si[t] = bi;
    __syncthreads();
    for (int st = 128; st > 0; st >>= 1) {
        if (t < st) {
            if (sb[t + st] > sb[t] || (sb[t + st] == sb[t] && si[t + st] < si[t])) {
                sb[t] = sb[t + st]; si[t] = si[t + st];
            }
        }
        __syncthreads();
    }
    if (t == 0) { pScore[blockIdx.x] = sb[0]; pIdx[blockIdx.x] = si[0]; }
}

__global__ __launch_bounds__(256)
void argmax_final(const float* __restrict__ pScore, const int* __restrict__ pIdx,
                  int* __restrict__ ridx) {
    __shared__ float sb[256]; __shared__ int si[256];
    int t = threadIdx.x;
    sb[t] = pScore[t]; si[t] = pIdx[t];
    __syncthreads();
    for (int st = 128; st > 0; st >>= 1) {
        if (t < st) {
            if (sb[t + st] > sb[t] || (sb[t + st] == sb[t] && si[t + st] < si[t])) {
                sb[t] = sb[t + st]; si[t] = si[t + st];
            }
        }
        __syncthreads();
    }
    if (t == 0) ridx[0] = si[0];
}

__global__ __launch_bounds__(256)
void write_out(const float* __restrict__ QOUT, const float* __restrict__ mask,
               const int* __restrict__ ridx, float* __restrict__ out) {
    int i = blockIdx.x * blockDim.x + threadIdx.x;
    if (i >= N_NODES) return;
    int ra = ridx[0];
    float m  = mask[i];
    float m2 = (i == ra) ? 0.0f : m;
    out[(size_t)i * 2]     = QOUT[(size_t)i * 2]     + rm_of(m, i);
    out[(size_t)i * 2 + 1] = QOUT[(size_t)i * 2 + 1] + m2;
}

// ---------------------------------------------------------------------------
// Host orchestration
// ---------------------------------------------------------------------------
extern "C" void kernel_launch(void* const* d_in, const int* in_sizes, int n_in,
                              void* d_out, int out_size, void* d_ws, size_t ws_size,
                              hipStream_t stream) {
    (void)in_sizes; (void)n_in; (void)out_size; (void)ws_size;
    const int N = N_NODES, E = N_EDGES;

    const float* node_type = (const float*)d_in[0];
    const float* requests  = (const float*)d_in[1];
    const float* upd_step  = (const float*)d_in[2];
    const float* latency   = (const float*)d_in[3];
    const float* amask     = (const float*)d_in[4];
    const int*   eidx      = (const int*)d_in[5];

    // workspace carve-up (256-byte aligned)
    char* ws = (char*)d_ws;
    size_t off = 0;
    auto alloc = [&](size_t bytes) { size_t o = off; off += (bytes + 255) & ~(size_t)255; return o; };
    bf16*     XBF   = (bf16*)(ws + alloc((size_t)N * 32 * 2));
    bf16*     COMB  = (bf16*)(ws + alloc((size_t)N * 128 * 2));   // also head h1
    bf16*     H2    = (bf16*)(ws + alloc((size_t)N * 64 * 2));
    bf16*     H3    = (bf16*)(ws + alloc((size_t)N * 32 * 2));
    float*    QKVS  = (float*)(ws + alloc((size_t)N * 512 * 4));
    float*    AGG   = (float*)(ws + alloc((size_t)N * 128 * 4));
    float*    XF32  = (float*)(ws + alloc((size_t)N * 32 * 4));
    float*    ALPHA = (float*)(ws + alloc((size_t)E * 4 * 4));
    unsigned* AMAXU = (unsigned*)(ws + alloc((size_t)N * 4 * 4));
    float*    DEN   = (float*)(ws + alloc((size_t)N * 4 * 4));
    float*    QOUT  = (float*)(ws + alloc((size_t)N * 2 * 4));
    float*    BIASC = (float*)(ws + alloc(4 * 512 * 4));
    float*    BNST  = (float*)(ws + alloc(64 * 4));
    float*    PSC   = (float*)(ws + alloc(256 * 4));
    int*      PIX   = (int*)(ws + alloc(256 * 4));
    int*      RIDX  = (int*)(ws + alloc(16));
    bf16*     BPACK = (bf16*)(ws + alloc((size_t)97024 * 2));

    // packed-weight sub-offsets (bf16 elements)
    bf16* BPQ[4], *BPT[4];
    for (int l = 0; l < 4; ++l) BPQ[l] = BPACK + (size_t)l * 16384;           // 32 frags
    for (int l = 0; l < 4; ++l) BPT[l] = BPACK + 65536 + (size_t)l * 4096;    // 8 frags
    bf16* BPH1 = BPACK + 81920;   // 8 frags
    bf16* BPH2 = BPACK + 86016;   // 16 frags
    bf16* BPH3 = BPACK + 94208;   // 4 frags
    bf16* BPH4 = BPACK + 96256;   // 1 frag

    auto pack = [&](const float* W, bf16* Bp, int ktC, int ntBase, int ntC,
                    int ntStride, int aK, int rowlen) {
        int total = ktC * ntC * 512;
        pack_b<<<(total + 255) / 256, 256, 0, stream>>>(W, Bp, ktC, ntBase, ntC,
                                                        ntStride, aK, rowlen);
    };

    // pack conv-layer weights (Wq,Wk,Wv,Ws -> 512-wide fused B), transf, head
    for (int l = 0; l < 4; ++l) {
        int base = 6 + 11 * l;
        int aK = (l == 0) ? 3 : 32;
        const int widx[4] = {0, 2, 4, 8};   // Wq, Wk, Wv, Ws
        for (int m = 0; m < 4; ++m)
            pack((const float*)d_in[base + widx[m]], BPQ[l], 1, 8 * m, 8, 32, aK, 128);
        concat_bias<<<2, 256, 0, stream>>>((const float*)d_in[base + 1],
                                           (const float*)d_in[base + 3],
                                           (const float*)d_in[base + 5],
                                           (const float*)d_in[base + 9],
                                           BIASC + l * 512);
        pack((const float*)d_in[50 + 2 * l], BPT[l], 4, 0, 2, 2, 128, 32);
    }
    pack((const float*)d_in[66], BPH1, 1, 0, 8, 8, 32, 128);
    pack((const float*)d_in[68], BPH2, 4, 0, 4, 4, 128, 64);
    pack((const float*)d_in[70], BPH3, 2, 0, 2, 2, 64, 32);
    pack((const float*)d_in[72], BPH4, 1, 0, 1, 1, 32, 2);

    auto gemm = [&](const bf16* A, int K, const bf16* Bp, const float* bias,
                    float* Cf, bf16* Cb, int NC, int storeCols, int relu) {
        int tiles = (N >> 4) * (NC >> 4);
        int thr = tiles * 32;
        gemm_wmma<<<(thr + 255) / 256, 256, 0, stream>>>(A, K, Bp, bias, Cf, Cb,
                                                         N, NC, storeCols, relu);
    };

    // layer-0 input
    build_x0<<<(N * 32 + 255) / 256, 256, 0, stream>>>(node_type, requests, upd_step, XBF);

    int egrid = (E * 4 + 255) / 256;
    for (int l = 0; l < 4; ++l) {
        int base = 6 + 11 * l;
        const float* We = (const float*)d_in[base + 6];
        const float* be = (const float*)d_in[base + 7];
        const float* Wb = (const float*)d_in[base + 10];
        const float* bt = (const float*)d_in[50 + 2 * l + 1];
        const float* g  = (const float*)d_in[58 + 2 * l];
        const float* bb = (const float*)d_in[58 + 2 * l + 1];

        // q|k|v|s = x @ [Wq|Wk|Wv|Ws] + bias  (single fused WMMA GEMM, K=32)
        gemm(XBF, 32, BPQ[l], BIASC + l * 512, QKVS, nullptr, 512, 512, 0);

        hipMemsetAsync(AGG,   0, (size_t)N * 128 * 4, stream);
        hipMemsetAsync(AMAXU, 0, (size_t)N * 4 * 4,   stream);  // fmap-identity
        hipMemsetAsync(DEN,   0, (size_t)N * 4 * 4,   stream);

        edge_alpha<<<egrid, 256, 0, stream>>>(eidx, latency, QKVS, We, be, ALPHA, AMAXU);
        edge_exp  <<<egrid, 256, 0, stream>>>(eidx, AMAXU, ALPHA, DEN);
        edge_msg  <<<egrid, 256, 0, stream>>>(eidx, latency, QKVS, We, be, ALPHA, DEN, AGG);

        gate<<<(N + 255) / 256, 256, 0, stream>>>(QKVS, AGG, Wb, COMB);

        // transform 128 -> 32, relu
        gemm(COMB, 128, BPT[l], bt, XF32, nullptr, 32, 32, 1);

        bn_stats<<<32, 256, 0, stream>>>(XF32, BNST);
        bn_apply<<<(N * 32 + 255) / 256, 256, 0, stream>>>(XF32, BNST, g, bb, XBF);
    }

    // head MLP: 32 -> 128 -> 64 -> 32 -> 2
    gemm(XBF,  32,  BPH1, (const float*)d_in[67], nullptr, COMB, 128, 128, 1);
    gemm(COMB, 128, BPH2, (const float*)d_in[69], nullptr, H2,   64,  64,  1);
    gemm(H2,   64,  BPH3, (const float*)d_in[71], nullptr, H3,   32,  32,  1);
    gemm(H3,   32,  BPH4, (const float*)d_in[73], QOUT,    nullptr, 16, 2, 0);

    // epilogue: argmax + masked output
    argmax_part <<<256, 256, 0, stream>>>(QOUT, amask, PSC, PIX);
    argmax_final<<<1,   256, 0, stream>>>(PSC, PIX, RIDX);
    write_out   <<<(N + 255) / 256, 256, 0, stream>>>(QOUT, amask, RIDX, (float*)d_out);
}
